// myModel_66709432042167
// MI455X (gfx1250) — matrix operand order repaired and checked by reading, not compiled, not run on hardware
//
#include <hip/hip_runtime.h>

typedef __attribute__((ext_vector_type(16))) _Float16 v16h;
typedef __attribute__((ext_vector_type(8)))  _Float16 v8h_t;
typedef __attribute__((ext_vector_type(8)))  float    v8f;

#define LDA  40   // 32 + 8 pad: rows 80B -> every fragment slice 16B-aligned
#define LDB  40   // B^T staging: 32 K + 8 pad
#define LDQ2 72   // 64 + 8 pad: rows 144B (16B-aligned)
#define LDV  40   // transposed 32-item tiles
#define LDT  40   // att tiles (32 K + 8 pad)

// ---------------- WMMA fragment helpers (CDNA5 wave32 layouts) ----------------
__device__ inline v16h cat8(v8h_t lo, v8h_t hi) {
  v16h r;
#pragma unroll
  for (int j = 0; j < 8; ++j) { r[j] = lo[j]; r[8 + j] = hi[j]; }
  return r;
}
// A (16x32 f16), row-major staging: lanes 0-15 row M=lane, K groups {0..7, 16..23};
//                                   lanes 16-31 row M=lane-16, K groups {8..15, 24..31}.
__device__ inline v16h load_A(const _Float16* S, int ld) {
  int lane = threadIdx.x & 31;
  int row = lane & 15;
  int kb = (lane < 16) ? 0 : 8;
  const _Float16* p = S + row * ld + kb;
  return cat8(*(const v8h_t*)p, *(const v8h_t*)(p + 16));
}
// B (32x16 f16) staged N-major (B^T): lane col N holds 16 K values in two runs of 8.
// lanes 0-15: col=lane, K groups {0..7, 16..23} ; lanes 16-31: col=lane-16, K groups {8..15, 24..31}.
__device__ inline v16h load_Bn(const _Float16* S, int ld) {
  int lane = threadIdx.x & 31;
  int col = lane & 15;
  int kb = (lane < 16) ? 0 : 8;
  const _Float16* p = S + col * ld + kb;
  return cat8(*(const v8h_t*)p, *(const v8h_t*)(p + 16));
}
__device__ inline v8f wmma32(v16h a, v16h b, v8f c) {
  return __builtin_amdgcn_wmma_f32_16x16x32_f16(false, a, false, b, (short)0, c, false, false);
}

// ---------------- generic f32 -> f16 cast ----------------
__global__ void cast_kernel(const float* __restrict__ in, _Float16* __restrict__ out, int n) {
  int i = blockIdx.x * blockDim.x + threadIdx.x;
  if (i < n) out[i] = (_Float16)in[i];
}

// ---------------- implicit-GEMM stride-2 3x3 SAME conv + bias + relu ----------------
// 4 waves/block share the B^T weight tile; each wave owns a 16-row im2col tile and
// computes NT 16-wide column tiles per A fragment. Row count must be a multiple of 64.
template <int NT, bool IN_F32>
__global__ __launch_bounds__(128) void conv2d_s2_wmma(
    const void* __restrict__ in_v, const _Float16* __restrict__ wgt,
    const float* __restrict__ bias, _Float16* __restrict__ out,
    int Hin, int Win, int Cin, int Cout, int Hout, int Wout, int Kreal) {
  const float* inF = (const float*)in_v;
  const _Float16* inH = (const _Float16*)in_v;
  __shared__ _Float16 As[4][16 * LDA];
  __shared__ _Float16 Bs[16 * NT * LDB];  // N-major (transposed) weight tile

  int wave = threadIdx.x >> 5;
  int lane = threadIdx.x & 31;
  int rowBase = (blockIdx.x * 4 + wave) * 16;
  int cb = blockIdx.y * (16 * NT);
  int Kpad = (Kreal + 31) & ~31;

  v8f acc[NT] = {};
  for (int kc = 0; kc < Kpad; kc += 32) {
    if (kc + 32 < Kpad)
      __builtin_prefetch(&wgt[(kc + 32) * Cout + cb], 0, 1);
    // B^T tile: 16*NT cols x 32 K, all 128 threads cooperate (coalesced over n)
#pragma unroll
    for (int t = 0; t < NT * 4; ++t) {
      int e = t * 128 + threadIdx.x;
      int n = e % (16 * NT);
      int kk = e / (16 * NT);
      int kg = kc + kk;
      _Float16 w = (_Float16)0.f;
      if (kg < Kreal) w = wgt[kg * Cout + cb + n];
      Bs[n * LDB + kk] = w;
    }
    // A tile (im2col gather): lane owns K-column kc+lane
    int kg = kc + lane;
    bool kvalid = kg < Kreal;
    int ci = 0, kx = 0, ky = 0;
    if (kvalid) { ci = kg % Cin; int t2 = kg / Cin; kx = t2 % 3; ky = t2 / 3; }
#pragma unroll 4
    for (int r = 0; r < 16; ++r) {
      int rg = rowBase + r;
      _Float16 val = (_Float16)0.f;
      if (kvalid) {
        int img = rg / (Hout * Wout);
        int rem = rg - img * Hout * Wout;
        int oy = rem / Wout;
        int ox = rem - oy * Wout;
        int iy = oy * 2 + ky;  // SAME pad, even input, stride 2: pad_lo = 0
        int ix = ox * 2 + kx;
        if (iy < Hin && ix < Win) {
          long idx = (((long)img * Hin + iy) * Win + ix) * Cin + ci;
          if (IN_F32) val = (_Float16)inF[idx]; else val = inH[idx];
        }
      }
      As[wave][r * LDA + lane] = val;
    }
    __syncthreads();
    v16h a = load_A(&As[wave][0], LDA);
#pragma unroll
    for (int nt = 0; nt < NT; ++nt) {
      v16h b = load_Bn(&Bs[nt * 16 * LDB], LDB);
      acc[nt] = wmma32(a, b, acc[nt]);
    }
    __syncthreads();
  }
#pragma unroll
  for (int nt = 0; nt < NT; ++nt) {
    int col = cb + nt * 16 + (lane & 15);
    float bsv = bias[col];
#pragma unroll
    for (int v = 0; v < 8; ++v) {
      int m = (lane < 16) ? v : v + 8;
      float x = acc[nt][v] + bsv;
      out[(long)(rowBase + m) * Cout + col] = (_Float16)(x > 0.f ? x : 0.f);
    }
  }
}

// ---------------- generic f16 GEMM: C = scale * (A[M,K] @ B[K,N]) ----------------
// M % 64 == 0, K % 32 == 0, grid = (M/64, N/(16*NT))
template <int NT>
__global__ __launch_bounds__(128) void gemm_f16_wmma(
    const _Float16* __restrict__ A, const _Float16* __restrict__ B,
    _Float16* __restrict__ C, int M, int N, int K, float scale) {
  __shared__ _Float16 As[4][16 * LDA];
  __shared__ _Float16 Bs[16 * NT * LDB];
  int wave = threadIdx.x >> 5, lane = threadIdx.x & 31;
  int rowBase = (blockIdx.x * 4 + wave) * 16;
  int cb = blockIdx.y * (16 * NT);
  v8f acc[NT] = {};
  for (int kc = 0; kc < K; kc += 32) {
#pragma unroll
    for (int t = 0; t < NT * 4; ++t) {
      int e = t * 128 + threadIdx.x;
      int n = e % (16 * NT);
      int kk = e / (16 * NT);
      Bs[n * LDB + kk] = B[(kc + kk) * N + cb + n];
    }
#pragma unroll 4
    for (int r = 0; r < 16; ++r)
      As[wave][r * LDA + lane] = A[(long)(rowBase + r) * K + kc + lane];
    __syncthreads();
    v16h a = load_A(&As[wave][0], LDA);
#pragma unroll
    for (int nt = 0; nt < NT; ++nt) {
      v16h b = load_Bn(&Bs[nt * 16 * LDB], LDB);
      acc[nt] = wmma32(a, b, acc[nt]);
    }
    __syncthreads();
  }
#pragma unroll
  for (int nt = 0; nt < NT; ++nt) {
    int col = cb + nt * 16 + (lane & 15);
#pragma unroll
    for (int v = 0; v < 8; ++v) {
      int m = (lane < 16) ? v : v + 8;
      C[(long)(rowBase + m) * N + col] = (_Float16)(acc[nt][v] * scale);
    }
  }
}

// ---------------- global average pool: [2048, 8*8, 64] -> [2048, 64] ----------------
__global__ void gap_kernel(const _Float16* __restrict__ in, _Float16* __restrict__ out) {
  int n = blockIdx.x, c = threadIdx.x;
  float s = 0.f;
  for (int p = 0; p < 64; ++p) s += (float)in[((long)n * 64 + p) * 64 + c];
  out[n * 64 + c] = (_Float16)(s * (1.f / 64.f));
}

// ---------------- encoder: block-diagonal linear attention per set ----------------
// enc[s*32+m, :] = sum_h cw[h]/256 * (Q_s,h K_s,h^T) V_s,h  (one wave per set)
__global__ __launch_bounds__(32) void enc_attention(
    const _Float16* __restrict__ q, const _Float16* __restrict__ k,
    const _Float16* __restrict__ v, const float* __restrict__ cw,
    _Float16* __restrict__ enc) {
  __shared__ _Float16 Qs[32 * LDQ2];
  __shared__ _Float16 Ks[32 * LDQ2];       // row-major: doubles as N-major K^T tile
  __shared__ _Float16 VsT[2][64 * LDV];    // transposed: VsT[d][item]
  __shared__ _Float16 Ts[2][32 * LDT];     // att, row-major
  int lane = threadIdx.x;
  int s = blockIdx.x;
  for (int h = 0; h < 2; ++h) {
    int base = (s * 32 + lane) * 128 + h * 64;
    const v8h_t* qsrc = (const v8h_t*)(q + base);
    const v8h_t* ksrc = (const v8h_t*)(k + base);
    const v8h_t* vsrc = (const v8h_t*)(v + base);
#pragma unroll
    for (int t = 0; t < 8; ++t) {
      *(v8h_t*)&Qs[lane * LDQ2 + t * 8] = qsrc[t];
      *(v8h_t*)&Ks[lane * LDQ2 + t * 8] = ksrc[t];
      v8h_t vv = vsrc[t];
#pragma unroll
      for (int j = 0; j < 8; ++j) VsT[h][(t * 8 + j) * LDV + lane] = vv[j];
    }
    __syncthreads();
    float scale = cw[h] * (1.f / 256.f);  // 1/sqrt(64) * 1/NITEM
    for (int mt = 0; mt < 2; ++mt)
      for (int nt = 0; nt < 2; ++nt) {
        v8f acc = {};
#pragma unroll
        for (int kc = 0; kc < 64; kc += 32) {
          v16h a = load_A(&Qs[mt * 16 * LDQ2 + kc], LDQ2);
          v16h b = load_Bn(&Ks[nt * 16 * LDQ2 + kc], LDQ2);  // rows of K == cols of K^T
          acc = wmma32(a, b, acc);
        }
        int col = nt * 16 + (lane & 15);
#pragma unroll
        for (int vv = 0; vv < 8; ++vv) {
          int m = mt * 16 + ((lane < 16) ? vv : vv + 8);
          Ts[h][m * LDT + col] = (_Float16)(acc[vv] * scale);
        }
      }
    __syncthreads();
  }
  for (int mt = 0; mt < 2; ++mt)
    for (int nt = 0; nt < 4; ++nt) {
      v8f acc = {};
#pragma unroll
      for (int h = 0; h < 2; ++h) {
        v16h a = load_A(&Ts[h][mt * 16 * LDT], LDT);      // K = 32 items
        v16h b = load_Bn(&VsT[h][nt * 16 * LDV], LDV);
        acc = wmma32(a, b, acc);
      }
      int col = nt * 16 + (lane & 15);
#pragma unroll
      for (int vv = 0; vv < 8; ++vv) {
        int m = mt * 16 + ((lane < 16) ? vv : vv + 8);
        enc[(s * 32 + m) * 64 + col] = (_Float16)acc[vv];
      }
    }
}

// ---------------- per-set enc sum: Esum[j,d] = sum_a enc[j*32+a, d] ----------------
__global__ void esum_kernel(const _Float16* __restrict__ enc, _Float16* __restrict__ esum) {
  int j = blockIdx.x, d = threadIdx.x;
  float s = 0.f;
  for (int a = 0; a < 32; ++a) s += (float)enc[(j * 32 + a) * 64 + d];
  esum[j * 64 + d] = (_Float16)s;
}

// ---------------- M[h,s] = (cw[h]/256) * K_s^T V_s  (64x64, one wave per (s,h)) ----------------
__global__ __launch_bounds__(32) void kv_outer(
    const _Float16* __restrict__ k, const _Float16* __restrict__ v,
    const float* __restrict__ cw, _Float16* __restrict__ Mout) {
  __shared__ _Float16 KsT[64 * LDV];  // KsT[d][item]: row-major A for K^T
  __shared__ _Float16 VsT[64 * LDV];  // VsT[e][item]: N-major B
  int lane = threadIdx.x;
  int s = blockIdx.x, h = blockIdx.y;
  int base = (s * 32 + lane) * 128 + h * 64;
  const v8h_t* ksrc = (const v8h_t*)(k + base);
  const v8h_t* vsrc = (const v8h_t*)(v + base);
#pragma unroll
  for (int t = 0; t < 8; ++t) {
    v8h_t kk = ksrc[t], vv = vsrc[t];
#pragma unroll
    for (int j = 0; j < 8; ++j) {
      KsT[(t * 8 + j) * LDV + lane] = kk[j];
      VsT[(t * 8 + j) * LDV + lane] = vv[j];
    }
  }
  __syncthreads();
  float scale = cw[h] * (1.f / 256.f);
  for (int mt = 0; mt < 4; ++mt)
    for (int nt = 0; nt < 4; ++nt) {
      v8f acc = {};
      v16h a = load_A(&KsT[mt * 16 * LDV], LDV);  // A[d][j] = K[j][d], K-dim = 32 items
      v16h b = load_Bn(&VsT[nt * 16 * LDV], LDV);
      acc = wmma32(a, b, acc);
      int col = nt * 16 + (lane & 15);
#pragma unroll
      for (int vv = 0; vv < 8; ++vv) {
        int m = mt * 16 + ((lane < 16) ? vv : vv + 8);
        Mout[(((long)h * 64 + s) * 64 + m) * 64 + col] = (_Float16)(acc[vv] * scale);
      }
    }
}

// ---------------- D[j,i,:] = sum_h Qsum_h[j,:] @ M[h,i]  (one wave per i) ----------------
__global__ __launch_bounds__(32) void d_kernel(
    const _Float16* __restrict__ qsum, const _Float16* __restrict__ Mh,
    _Float16* __restrict__ D) {
  __shared__ _Float16 As[2][64 * LDQ2];   // qsum rows per head (row-major)
  __shared__ _Float16 BsT[2][64 * LDQ2];  // M^T per head (N-major: BsT[d][k])
  int lane = threadIdx.x, i = blockIdx.x;
  for (int h = 0; h < 2; ++h)
    for (int t = 0; t < 128; ++t) {
      int e = t * 32 + lane;
      int r = e >> 6, c = e & 63;
      As[h][r * LDQ2 + c] = qsum[r * 128 + h * 64 + c];
      BsT[h][c * LDQ2 + r] = Mh[(((long)h * 64 + i) * 64 + r) * 64 + c];
    }
  __syncthreads();
  for (int mt = 0; mt < 4; ++mt)
    for (int nt = 0; nt < 4; ++nt) {
      v8f acc = {};
#pragma unroll
      for (int h = 0; h < 2; ++h)
#pragma unroll
        for (int kc = 0; kc < 64; kc += 32) {
          v16h a = load_A(&As[h][mt * 16 * LDQ2 + kc], LDQ2);
          v16h b = load_Bn(&BsT[h][nt * 16 * LDQ2 + kc], LDQ2);
          acc = wmma32(a, b, acc);
        }
      int col = nt * 16 + (lane & 15);
#pragma unroll
      for (int vv = 0; vv < 8; ++vv) {
        int m = mt * 16 + ((lane < 16) ? vv : vv + 8);
        D[((long)m * 64 + i) * 64 + col] = (_Float16)acc[vv];
      }
    }
}

// ---------------- final scores + softmax ----------------
__global__ void score_kernel(const _Float16* __restrict__ S, const float* __restrict__ ccw,
                             const float* __restrict__ fcw, const float* __restrict__ fcb,
                             float* __restrict__ out) {
  int idx = blockIdx.x * blockDim.x + threadIdx.x;
  if (idx >= 4096) return;
  int j = idx >> 6, i = idx & 63;
  float score = 0.f;
  for (int h = 0; h < 2; ++h) {
    const _Float16* p1 = &S[((long)(j * 64 + i)) * 128 + h * 64];
    const _Float16* p2 = &S[((long)(i * 64 + j)) * 128 + h * 64];
    float dt = 0.f;
    for (int d = 0; d < 64; ++d) dt += (float)p1[d] * (float)p2[d];
    score += ccw[h] * dt;
  }
  score *= (1.f / (8.f * 1024.f));  // /sqrt(64) /NITEM^2
  float l0 = score * fcw[0] + fcb[0];
  float l1 = score * fcw[1] + fcb[1];
  float m = fmaxf(l0, l1);
  float e0 = __expf(l0 - m), e1 = __expf(l1 - m);
  float inv = 1.f / (e0 + e1);
  out[idx * 2 + 0] = e0 * inv;
  out[idx * 2 + 1] = e1 * inv;
}

// =====================================================================
extern "C" void kernel_launch(void* const* d_in, const int* in_sizes, int n_in,
                              void* d_out, int out_size, void* d_ws, size_t ws_size,
                              hipStream_t stream) {
  (void)in_sizes; (void)n_in; (void)out_size; (void)ws_size;
  const float* x   = (const float*)d_in[0];
  const float* c1w = (const float*)d_in[1];  const float* c1b = (const float*)d_in[2];
  const float* c2w = (const float*)d_in[3];  const float* c2b = (const float*)d_in[4];
  const float* c3w = (const float*)d_in[5];  const float* c3b = (const float*)d_in[6];
  const float* ew1 = (const float*)d_in[7];  const float* ew2 = (const float*)d_in[8];
  const float* ew3 = (const float*)d_in[9];  const float* ecw = (const float*)d_in[10];
  const float* dw1 = (const float*)d_in[11]; const float* dw2 = (const float*)d_in[12];
  const float* dw3 = (const float*)d_in[13]; const float* dcw = (const float*)d_in[14];
  const float* csw = (const float*)d_in[15]; const float* ccw = (const float*)d_in[16];
  const float* fcw = (const float*)d_in[17]; const float* fcb = (const float*)d_in[18];

  char* ws = (char*)d_ws;
  size_t off = 0;
  auto alloc = [&](size_t elems) {
    size_t o = off;
    off += (elems * sizeof(_Float16) + 255) & ~(size_t)255;
    return (_Float16*)(ws + o);
  };
  // weights (f16, persistent)
  _Float16* w1h  = alloc(864);
  _Float16* w2h  = alloc(18432);
  _Float16* w3h  = alloc(36864);
  _Float16* ew1h = alloc(8192);
  _Float16* ew2h = alloc(8192);
  _Float16* ew3h = alloc(8192);
  _Float16* dw1h = alloc(8192);
  _Float16* dw2h = alloc(8192);
  _Float16* dw3h = alloc(8192);
  _Float16* cswh = alloc(8192);
  // big activations: a1 dead after conv2, so post-conv3 buffers alias a1's region
  size_t a1_off = off;
  _Float16* a1 = alloc((size_t)2048 * 32 * 32 * 32);  // 134 MB
  _Float16* a2 = alloc((size_t)2048 * 16 * 16 * 64);  // 67 MB
  off = a1_off;                                       // reuse a1 region
  _Float16* a3     = alloc((size_t)2048 * 8 * 8 * 64);
  _Float16* feat   = alloc(2048 * 64);
  _Float16* qh     = alloc(2048 * 128);
  _Float16* kh     = alloc(2048 * 128);
  _Float16* vh     = alloc(2048 * 128);
  _Float16* enc16  = alloc(2048 * 64);
  _Float16* esum16 = alloc(64 * 64);
  _Float16* qsum16 = alloc(64 * 128);
  _Float16* dk16   = alloc(2048 * 128);
  _Float16* dv16   = alloc(2048 * 128);
  _Float16* Mh16   = alloc(2 * 64 * 64 * 64);
  _Float16* D16    = alloc(64 * 64 * 64);
  _Float16* S16    = alloc(4096 * 128);

  auto cast = [&](const float* s, _Float16* d, int n) {
    cast_kernel<<<(n + 255) / 256, 256, 0, stream>>>(s, d, n);
  };
  cast(c1w, w1h, 864);   cast(c2w, w2h, 18432); cast(c3w, w3h, 36864);
  cast(ew1, ew1h, 8192); cast(ew2, ew2h, 8192); cast(ew3, ew3h, 8192);
  cast(dw1, dw1h, 8192); cast(dw2, dw2h, 8192); cast(dw3, dw3h, 8192);
  cast(csw, cswh, 8192);

  // convs (implicit GEMM, stride 2, SAME); row counts all multiples of 64
  conv2d_s2_wmma<2, true><<<dim3(2097152 / 64, 1), 128, 0, stream>>>(
      x, w1h, c1b, a1, 64, 64, 3, 32, 32, 32, 27);
  conv2d_s2_wmma<4, false><<<dim3(524288 / 64, 1), 128, 0, stream>>>(
      a1, w2h, c2b, a2, 32, 32, 32, 64, 16, 16, 288);
  conv2d_s2_wmma<4, false><<<dim3(131072 / 64, 1), 128, 0, stream>>>(
      a2, w3h, c3b, a3, 16, 16, 64, 64, 8, 8, 576);
  gap_kernel<<<2048, 64, 0, stream>>>(a3, feat);

  auto gemmN128 = [&](const _Float16* A, const _Float16* B, _Float16* C, int M, int K) {
    gemm_f16_wmma<8><<<dim3(M / 64, 1), 128, 0, stream>>>(A, B, C, M, 128, K, 1.f);
  };

  // encoder QKV + block-diagonal attention
  gemmN128(feat, ew1h, qh, 2048, 64);
  gemmN128(feat, ew2h, kh, 2048, 64);
  gemmN128(feat, ew3h, vh, 2048, 64);
  enc_attention<<<64, 32, 0, stream>>>(qh, kh, vh, ecw, enc16);

  // decoder collapsed by linearity: Qsum = (sum_a enc) @ dec_w1 ; M[h,s] = K_s^T V_s
  esum_kernel<<<64, 64, 0, stream>>>(enc16, esum16);
  gemmN128(esum16, dw1h, qsum16, 64, 64);
  gemmN128(enc16, dw2h, dk16, 2048, 64);
  gemmN128(enc16, dw3h, dv16, 2048, 64);
  kv_outer<<<dim3(64, 2), 32, 0, stream>>>(dk16, dv16, dcw, Mh16);
  d_kernel<<<64, 32, 0, stream>>>(qsum16, Mh16, D16);

  // css projection + final scores/softmax
  gemmN128(D16, cswh, S16, 4096, 64);
  score_kernel<<<16, 256, 0, stream>>>(S16, ccw, fcw, fcb, (float*)d_out);
}
